// PatchGCN_8426725834976
// MI455X (gfx1250) — matrix operand order repaired and checked by reading, not compiled, not run on hardware
//
#include <hip/hip_runtime.h>
#include <hip/hip_bf16.h>
#include <math.h>

// ---------------------------------------------------------------------------
// Types for CDNA5 WMMA
// ---------------------------------------------------------------------------
typedef __attribute__((ext_vector_type(8)))  float  v8f;
typedef __attribute__((ext_vector_type(16))) __bf16 v16bf;

union BFrag { v16bf v; unsigned u[8]; };

__device__ __forceinline__ unsigned short f2bf(float f) {
    unsigned u = __builtin_bit_cast(unsigned, f);
    u += 0x7FFFu + ((u >> 16) & 1u);          // round-to-nearest-even
    return (unsigned short)(u >> 16);
}

__device__ __forceinline__ unsigned enc_ord(float f) {
    unsigned b = __builtin_bit_cast(unsigned, f);
    return (b & 0x80000000u) ? ~b : (b | 0x80000000u);
}
__device__ __forceinline__ float dec_ord(unsigned e) {
    unsigned b = (e & 0x80000000u) ? (e & 0x7FFFFFFFu) : ~e;
    return __builtin_bit_cast(float, b);
}

// ---------------------------------------------------------------------------
// Generic WMMA GEMM:  C[M,N] = act(A[M,K] @ W[K,N] + bias[N])
// BLOCK: 128 rows x 128 cols, 256 threads (8 wave32 waves). grid.y = N/128.
// Double-buffered LDS tiles; global loads for tile i+1 issued before the
// WMMAs of tile i (register double buffering); B fragments preloaded in
// groups of 4 so WMMA overlaps ds_load latency.
// ACT: 0=none 1=relu 2=tanh 3=sigmoid
// ---------------------------------------------------------------------------
template <int ACT>
__global__ __launch_bounds__(256) void gemm_wmma(
    const float* __restrict__ A, int lda,
    const float* __restrict__ W, int ldw,
    const float* __restrict__ bias,
    float* __restrict__ C, int ldc,
    int M, int K)
{
    constexpr int KP = 40;                            // padded K stride (elems)
    __shared__ __align__(16) unsigned short As[2][128 * KP];
    __shared__ __align__(16) unsigned short Bs[2][128 * KP];  // transposed [n][k]

    const int t    = threadIdx.x;
    const int lane = t & 31;
    const int wv   = t >> 5;
    const int mbase = blockIdx.x * 128;
    const int nbase = blockIdx.y * 128;

    // cooperative-load coordinates
    const int arr = t >> 3;          // A row within tile  (0..31, +32 per pass)
    const int akk = (t & 7) * 4;     // A k within tile
    const int bkk = t >> 5;          // B k within tile    (0..7, +8 per pass)
    const int bnn = (t & 31) * 4;    // B col within tile

    v8f acc[8];
    #pragma unroll
    for (int j = 0; j < 8; ++j)
        #pragma unroll
        for (int r = 0; r < 8; ++r) acc[j][r] = 0.0f;

    float4 ra[4], rb[4];
    float  amsk[4];

    auto gload = [&](int k0) {
        // streaming-A prefetch hint one tile ahead
        {
            int g0 = mbase + arr;
            int c0 = g0 < M ? g0 : M - 1;
            __builtin_prefetch(A + (size_t)c0 * lda + k0 + 32 + akk, 0, 0);
        }
        #pragma unroll
        for (int p = 0; p < 4; ++p) {
            int grow = mbase + arr + p * 32;
            int crow = grow < M ? grow : M - 1;       // branchless clamp
            amsk[p]  = grow < M ? 1.0f : 0.0f;
            ra[p] = *(const float4*)(A + (size_t)crow * lda + k0 + akk);
            rb[p] = *(const float4*)(W + (size_t)(k0 + bkk + p * 8) * ldw + nbase + bnn);
        }
    };

    auto sstore = [&](int buf) {
        #pragma unroll
        for (int p = 0; p < 4; ++p) {
            float m = amsk[p];
            unsigned lo = (unsigned)f2bf(ra[p].x * m) | ((unsigned)f2bf(ra[p].y * m) << 16);
            unsigned hi = (unsigned)f2bf(ra[p].z * m) | ((unsigned)f2bf(ra[p].w * m) << 16);
            *(unsigned*)&As[buf][(arr + p * 32) * KP + akk + 0] = lo;
            *(unsigned*)&As[buf][(arr + p * 32) * KP + akk + 2] = hi;
            int kk = bkk + p * 8;
            Bs[buf][(bnn + 0) * KP + kk] = f2bf(rb[p].x);
            Bs[buf][(bnn + 1) * KP + kk] = f2bf(rb[p].y);
            Bs[buf][(bnn + 2) * KP + kk] = f2bf(rb[p].z);
            Bs[buf][(bnn + 3) * KP + kk] = f2bf(rb[p].w);
        }
    };

    const int arow = wv * 16 + (lane & 15);
    const int kh   = lane >> 4;

    auto compute = [&](int buf) {
        // A frag per ISA layout: e0..7 -> K=8h+e ; e8..15 -> K=16+8h+(e-8)
        BFrag a;
        *(uint4*)&a.u[0] = *(const uint4*)&As[buf][arow * KP + kh * 8];
        *(uint4*)&a.u[4] = *(const uint4*)&As[buf][arow * KP + 16 + kh * 8];
        // B frags in groups of 4: loads issued ahead of WMMAs -> overlap
        BFrag b[4];
        #pragma unroll
        for (int j = 0; j < 4; ++j) {
            const uint4* bp = (const uint4*)&Bs[buf][(j * 16 + (lane & 15)) * KP + kh * 8];
            *(uint4*)&b[j].u[0] = bp[0];
            *(uint4*)&b[j].u[4] = bp[2];
        }
        #pragma unroll
        for (int j = 0; j < 4; ++j)
            acc[j] = __builtin_amdgcn_wmma_f32_16x16x32_bf16(
                false, a.v, false, b[j].v, (short)0, acc[j], false, false);
        #pragma unroll
        for (int j = 0; j < 4; ++j) {
            const uint4* bp = (const uint4*)&Bs[buf][((j + 4) * 16 + (lane & 15)) * KP + kh * 8];
            *(uint4*)&b[j].u[0] = bp[0];
            *(uint4*)&b[j].u[4] = bp[2];
        }
        #pragma unroll
        for (int j = 0; j < 4; ++j)
            acc[j + 4] = __builtin_amdgcn_wmma_f32_16x16x32_bf16(
                false, a.v, false, b[j].v, (short)0, acc[j + 4], false, false);
    };

    const int NT = K / 32;
    gload(0);
    sstore(0);
    __syncthreads();
    for (int i = 0; i < NT; ++i) {
        if (i + 1 < NT) gload((i + 1) * 32);   // global loads in flight ...
        compute(i & 1);                        // ... while WMMAs run
        if (i + 1 < NT) {
            sstore((i + 1) & 1);
            __syncthreads();
        }
    }

    // ---- epilogue: D layout: vgpr r, lanes<16: M=r, lanes>=16: M=r+8 -------
    const int colb = lane & 15;
    const int rowh = lane >> 4;
    #pragma unroll
    for (int j = 0; j < 8; ++j) {
        int col = nbase + j * 16 + colb;
        float bv = bias[col];
        #pragma unroll
        for (int r = 0; r < 8; ++r) {
            int row = mbase + wv * 16 + rowh * 8 + r;
            if (row < M) {
                float v = acc[j][r] + bv;
                if (ACT == 1)      v = fmaxf(v, 0.0f);
                else if (ACT == 2) v = tanhf(v);
                else if (ACT == 3) v = 1.0f / (1.0f + __expf(-v));
                C[(size_t)row * ldc + col] = v;
            }
        }
    }
}

// ---------------------------------------------------------------------------
// Elementwise helpers
// ---------------------------------------------------------------------------
__global__ __launch_bounds__(256) void zero_k(float* __restrict__ p, long long n) {
    long long i = (long long)blockIdx.x * 256 + threadIdx.x;
    long long st = (long long)gridDim.x * 256;
    for (; i < n; i += st) p[i] = 0.0f;
}
__global__ __launch_bounds__(256) void copy_k(float* __restrict__ d,
                                              const float* __restrict__ s, long long n) {
    long long i = (long long)blockIdx.x * 256 + threadIdx.x;
    long long st = (long long)gridDim.x * 256;
    for (; i < n; i += st) d[i] = s[i];
}
// scatter a [M,128] buffer into a 128-col block of xcat (ld 512)
__global__ __launch_bounds__(256) void scatter_k(const float* __restrict__ s,
                                                 float* __restrict__ dc, long long n) {
    long long i = (long long)blockIdx.x * 256 + threadIdx.x;
    long long st = (long long)gridDim.x * 256;
    for (; i < n; i += st) dc[(i >> 7) * 512 + (i & 127)] = s[i];
}

// ---------------------------------------------------------------------------
// GENConv edge passes (one wave per edge, 4 channels per lane)
// ---------------------------------------------------------------------------
__global__ __launch_bounds__(256) void edge_max_k(
    const float* __restrict__ x, const int* __restrict__ src,
    const int* __restrict__ dst, unsigned* __restrict__ mx,
    const float* __restrict__ tptr, int E)
{
    int e = blockIdx.x * 8 + (threadIdx.x >> 5);
    if (e >= E) return;
    int lane = threadIdx.x & 31;
    float tt = *tptr;
    int s = src[e], d = dst[e];
    float4 v = *(const float4*)(x + (size_t)s * 128 + lane * 4);
    unsigned* mp = mx + (size_t)d * 128 + lane * 4;
    const float* vf = (const float*)&v;
    #pragma unroll
    for (int i = 0; i < 4; ++i) {
        float msg = fmaxf(vf[i], 0.0f) + 1e-7f;
        atomicMax(&mp[i], enc_ord(msg * tt));
    }
}

__global__ __launch_bounds__(256) void edge_sum_k(
    const float* __restrict__ x, const int* __restrict__ src,
    const int* __restrict__ dst, const unsigned* __restrict__ mx,
    float* __restrict__ ssum, float* __restrict__ nsum,
    const float* __restrict__ tptr, int E)
{
    int e = blockIdx.x * 8 + (threadIdx.x >> 5);
    if (e >= E) return;
    int lane = threadIdx.x & 31;
    float tt = *tptr;
    int s = src[e], d = dst[e];
    float4 v = *(const float4*)(x + (size_t)s * 128 + lane * 4);
    size_t base = (size_t)d * 128 + lane * 4;
    const float* vf = (const float*)&v;
    #pragma unroll
    for (int i = 0; i < 4; ++i) {
        float msg = fmaxf(vf[i], 0.0f) + 1e-7f;
        float m   = dec_ord(mx[base + i]);
        float ex  = __expf(msg * tt - m);
        __hip_atomic_fetch_add(&ssum[base + i], ex,
                               __ATOMIC_RELAXED, __HIP_MEMORY_SCOPE_AGENT);
        __hip_atomic_fetch_add(&nsum[base + i], msg * ex,
                               __ATOMIC_RELAXED, __HIP_MEMORY_SCOPE_AGENT);
    }
}

__global__ __launch_bounds__(256) void add_aggr_k(
    const float* __restrict__ x, const float* __restrict__ ssum,
    const float* __restrict__ nsum, float* __restrict__ out, long long n)
{
    long long i = (long long)blockIdx.x * 256 + threadIdx.x;
    long long st = (long long)gridDim.x * 256;
    for (; i < n; i += st) out[i] = x[i] + nsum[i] / (ssum[i] + 1e-16f);
}

// ---------------------------------------------------------------------------
// LayerNorm (+relu) — one wave per row
// ---------------------------------------------------------------------------
template <int C>
__global__ __launch_bounds__(256) void ln_relu_k(
    const float* __restrict__ g, const float* __restrict__ w,
    const float* __restrict__ b, float* __restrict__ out, int M)
{
    int row = blockIdx.x * 8 + (threadIdx.x >> 5);
    if (row >= M) return;
    int lane = threadIdx.x & 31;
    constexpr int PER = C / 32;
    float v[PER], s = 0.f, s2 = 0.f;
    #pragma unroll
    for (int i = 0; i < PER; ++i) {
        v[i] = g[(size_t)row * C + lane + i * 32];
        s += v[i]; s2 += v[i] * v[i];
    }
    #pragma unroll
    for (int o = 16; o > 0; o >>= 1) { s += __shfl_xor(s, o, 32); s2 += __shfl_xor(s2, o, 32); }
    float mu  = s / C;
    float inv = rsqrtf(s2 / C - mu * mu + 1e-5f);
    #pragma unroll
    for (int i = 0; i < PER; ++i) {
        int c = lane + i * 32;
        float y = (v[i] - mu) * inv * w[c] + b[c];
        out[(size_t)row * C + c] = fmaxf(y, 0.0f);
    }
}

// layers 1,2: x += relu(LN(o)); also write x into xcat block (ld 512)
__global__ __launch_bounds__(256) void post_layer_k(
    const float* __restrict__ o, const float* __restrict__ w,
    const float* __restrict__ b, float* __restrict__ xcur,
    float* __restrict__ xcat_off, int M)
{
    int row = blockIdx.x * 8 + (threadIdx.x >> 5);
    if (row >= M) return;
    int lane = threadIdx.x & 31;
    float v[4], s = 0.f, s2 = 0.f;
    #pragma unroll
    for (int i = 0; i < 4; ++i) {
        v[i] = o[(size_t)row * 128 + lane + i * 32];
        s += v[i]; s2 += v[i] * v[i];
    }
    #pragma unroll
    for (int t = 16; t > 0; t >>= 1) { s += __shfl_xor(s, t, 32); s2 += __shfl_xor(s2, t, 32); }
    float mu  = s / 128.f;
    float inv = rsqrtf(s2 / 128.f - mu * mu + 1e-5f);
    #pragma unroll
    for (int i = 0; i < 4; ++i) {
        int c = lane + i * 32;
        float h = fmaxf((v[i] - mu) * inv * w[c] + b[c], 0.0f);
        float xn = xcur[(size_t)row * 128 + c] + h;
        xcur[(size_t)row * 128 + c] = xn;
        xcat_off[(size_t)row * 512 + c] = xn;
    }
}

// ---------------------------------------------------------------------------
// Attention head
// ---------------------------------------------------------------------------
__global__ __launch_bounds__(256) void score_k(
    const float* __restrict__ a, const float* __restrict__ bm,
    const float* __restrict__ cw, const float* __restrict__ cb,
    float* __restrict__ esc, unsigned* __restrict__ gmx, int M)
{
    int row = blockIdx.x * 8 + (threadIdx.x >> 5);
    if (row >= M) return;
    int lane = threadIdx.x & 31;
    float s = 0.f;
    #pragma unroll
    for (int i = 0; i < 16; ++i) {
        int c = lane + i * 32;
        s += a[(size_t)row * 512 + c] * bm[(size_t)row * 512 + c] * cw[c];
    }
    #pragma unroll
    for (int o = 16; o > 0; o >>= 1) s += __shfl_xor(s, o, 32);
    if (lane == 0) {
        float v = s + cb[0];
        esc[row] = v;
        atomicMax(gmx, enc_ord(v));
    }
}

__global__ __launch_bounds__(256) void esum_k(
    float* __restrict__ esc, const unsigned* __restrict__ gmx,
    float* __restrict__ stot, int M)
{
    __shared__ float red[256];
    float m = dec_ord(*gmx);
    float local = 0.f;
    for (int i = blockIdx.x * 256 + threadIdx.x; i < M; i += gridDim.x * 256) {
        float e = __expf(esc[i] - m);
        esc[i] = e;
        local += e;
    }
    red[threadIdx.x] = local;
    __syncthreads();
    for (int o = 128; o > 0; o >>= 1) {
        if (threadIdx.x < o) red[threadIdx.x] += red[threadIdx.x + o];
        __syncthreads();
    }
    if (threadIdx.x == 0)
        __hip_atomic_fetch_add(stot, red[0], __ATOMIC_RELAXED, __HIP_MEMORY_SCOPE_AGENT);
}

// pool[c] = sum_n esc[n] * H[n][c] ; each block owns 64 channels
__global__ __launch_bounds__(256) void pool_k(
    const float* __restrict__ esc, const float* __restrict__ H,
    float* __restrict__ pool, int M)
{
    int c    = blockIdx.x * 64 + (threadIdx.x & 63);
    int nsub = threadIdx.x >> 6;   // 0..3
    float acc = 0.f;
    for (int n = nsub; n < M; n += 4) acc += esc[n] * H[(size_t)n * 512 + c];
    __shared__ float red[256];
    red[threadIdx.x] = acc;
    __syncthreads();
    if (nsub == 0)
        pool[c] = red[threadIdx.x] + red[threadIdx.x + 64] +
                  red[threadIdx.x + 128] + red[threadIdx.x + 192];
}

__global__ __launch_bounds__(256) void head_k(
    const float* __restrict__ pool, const float* __restrict__ stot,
    const float* __restrict__ rho_w, const float* __restrict__ rho_b,
    const float* __restrict__ cls_w, const float* __restrict__ cls_b,
    float* __restrict__ out)
{
    __shared__ float hp[512], r[512];
    int t = threadIdx.x;
    float inv = 1.0f / (*stot);
    hp[t] = pool[t] * inv;
    hp[t + 256] = pool[t + 256] * inv;
    __syncthreads();
    #pragma unroll
    for (int o = 0; o < 2; ++o) {
        int c = t + o * 256;
        float s = rho_b[c];
        for (int k = 0; k < 512; ++k) s += hp[k] * rho_w[(size_t)k * 512 + c];
        r[c] = fmaxf(s, 0.0f);
    }
    __syncthreads();
    if (t < 4) {
        float s = cls_b[t];
        for (int k = 0; k < 512; ++k) s += r[k] * cls_w[k * 4 + t];
        out[t] = s;
    }
}

// ---------------------------------------------------------------------------
// Host orchestration
// ---------------------------------------------------------------------------
extern "C" void kernel_launch(void* const* d_in, const int* in_sizes, int n_in,
                              void* d_out, int out_size, void* d_ws, size_t ws_size,
                              hipStream_t stream)
{
    const long long NN = 50000;
    const int E = in_sizes[1] / 2;

    const float* x       = (const float*)d_in[0];
    const int*   ei      = (const int*)d_in[1];
    const int*   src     = ei;
    const int*   dst     = ei + E;
    const float* fc_w    = (const float*)d_in[2];
    const float* fc_b    = (const float*)d_in[3];
    const float* conv_w1 = (const float*)d_in[4];
    const float* conv_b1 = (const float*)d_in[5];
    const float* conv_lnw= (const float*)d_in[6];
    const float* conv_lnb= (const float*)d_in[7];
    const float* conv_w2 = (const float*)d_in[8];
    const float* conv_b2 = (const float*)d_in[9];
    const float* conv_t  = (const float*)d_in[10];
    const float* norm_w  = (const float*)d_in[11];
    const float* norm_b  = (const float*)d_in[12];
    const float* phi_w   = (const float*)d_in[13];
    const float* phi_b   = (const float*)d_in[14];
    const float* aw      = (const float*)d_in[15];
    const float* ab      = (const float*)d_in[16];
    const float* bw      = (const float*)d_in[17];
    const float* bb      = (const float*)d_in[18];
    const float* cw      = (const float*)d_in[19];
    const float* cb      = (const float*)d_in[20];
    const float* rho_w   = (const float*)d_in[21];
    const float* rho_b   = (const float*)d_in[22];
    const float* cls_w   = (const float*)d_in[23];
    const float* cls_b   = (const float*)d_in[24];

    float* W8 = (float*)d_ws;
    float*    xcur = W8;
    float*    xcat = W8 + NN * 128;
    unsigned* maxb = (unsigned*)(W8 + NN * 640);
    float*    ssum = W8 + NN * 768;
    float*    nsum = W8 + NN * 896;
    float*    tmp  = W8 + NN * 1024;
    float*    g256 = W8 + NN * 1152;
    float*    h256 = W8 + NN * 1408;
    float*    o128 = W8 + NN * 1664;
    float*    H512 = W8 + NN * 1792;
    float*    amat = g256;              // reuse g256+h256 (512 cols)
    float*    bmat = (float*)maxb;      // reuse maxb..tmp   (512 cols)
    float*    esc  = W8 + NN * 2304;
    float*    pool = W8 + NN * 2305;
    float*    stot = pool + 512;
    unsigned* gmx  = (unsigned*)(stot + 1);

    const dim3 blk(256);
    const int MB = (int)((NN + 127) / 128);    // 391
    const int EW = 4096;                        // grid-stride elementwise blocks
    const int RW = (int)((NN + 7) / 8);        // one-wave-per-row grids
    const int EB = (E + 7) / 8;                // one-wave-per-edge grids

    // fc: x0 = relu(x @ fc_w + b)
    gemm_wmma<1><<<dim3(MB, 1), blk, 0, stream>>>(x, 1024, fc_w, 128, fc_b,
                                                  xcur, 128, (int)NN, 1024);
    scatter_k<<<EW, blk, 0, stream>>>(xcur, xcat + 0, NN * 128);

    for (int L = 0; L < 3; ++L) {
        zero_k<<<EW, blk, 0, stream>>>((float*)maxb, NN * 384);   // maxb+ssum+nsum
        edge_max_k<<<EB, blk, 0, stream>>>(xcur, src, dst, maxb, conv_t + L, E);
        edge_sum_k<<<EB, blk, 0, stream>>>(xcur, src, dst, maxb, ssum, nsum,
                                           conv_t + L, E);
        add_aggr_k<<<EW, blk, 0, stream>>>(xcur, ssum, nsum, tmp, NN * 128);
        gemm_wmma<0><<<dim3(MB, 2), blk, 0, stream>>>(tmp, 128,
            conv_w1 + (size_t)L * 128 * 256, 256, conv_b1 + (size_t)L * 256,
            g256, 256, (int)NN, 128);
        ln_relu_k<256><<<RW, blk, 0, stream>>>(g256, conv_lnw + (size_t)L * 256,
            conv_lnb + (size_t)L * 256, h256, (int)NN);
        gemm_wmma<0><<<dim3(MB, 1), blk, 0, stream>>>(h256, 256,
            conv_w2 + (size_t)L * 256 * 128, 128, conv_b2 + (size_t)L * 128,
            o128, 128, (int)NN, 256);
        if (L == 0) {
            copy_k<<<EW, blk, 0, stream>>>(xcur, o128, NN * 128);
            scatter_k<<<EW, blk, 0, stream>>>(o128, xcat + 128, NN * 128);
        } else {
            post_layer_k<<<RW, blk, 0, stream>>>(o128, norm_w + (size_t)L * 128,
                norm_b + (size_t)L * 128, xcur, xcat + 128 * (L + 1), (int)NN);
        }
    }

    // head
    gemm_wmma<1><<<dim3(MB, 4), blk, 0, stream>>>(xcat, 512, phi_w, 512, phi_b,
                                                  H512, 512, (int)NN, 512);
    gemm_wmma<2><<<dim3(MB, 4), blk, 0, stream>>>(H512, 512, aw, 512, ab,
                                                  amat, 512, (int)NN, 512);
    gemm_wmma<3><<<dim3(MB, 4), blk, 0, stream>>>(H512, 512, bw, 512, bb,
                                                  bmat, 512, (int)NN, 512);

    zero_k<<<1, blk, 0, stream>>>(pool, 514);   // pool[512], stot, gmx
    score_k<<<RW, blk, 0, stream>>>(amat, bmat, cw, cb, esc, gmx, (int)NN);
    esum_k<<<128, blk, 0, stream>>>(esc, gmx, stot, (int)NN);
    pool_k<<<8, blk, 0, stream>>>(esc, H512, pool, (int)NN);
    head_k<<<1, blk, 0, stream>>>(pool, stot, rho_w, rho_b, cls_w, cls_b,
                                  (float*)d_out);
}
